// TimeAwareGGNN_29403346108780
// MI455X (gfx1250) — matrix operand order repaired and checked by reading, not compiled, not run on hardware
//
#include <hip/hip_runtime.h>
#include <hip/hip_bf16.h>

// ---------------- problem constants (from reference) ----------------
#define N_NODES   102400      // B*L
#define N_B       512
#define N_L       200
#define N_H       128
#define N_E       819200
#define N_HEADS   4
#define N_DH      32
#define N_CLASSES 10000
#define E_DIM     64
#define KIN       96          // 66 input features padded to 96 (3 k-steps of 32)

typedef __attribute__((ext_vector_type(16))) __bf16 v16bf;
typedef __attribute__((ext_vector_type(8)))  __bf16 v8bf;
typedef __attribute__((ext_vector_type(8)))  float  v8f;

// ---------------- generic f32 -> bf16 convert ----------------
__global__ void k_cvt_bf16(const float* __restrict__ in, __bf16* __restrict__ out, int n) {
    int i = blockIdx.x * blockDim.x + threadIdx.x;
    if (i < n) out[i] = (__bf16)in[i];
}

// ---------------- pad input_w [128,66] -> bf16 [128,96] ----------------
__global__ void k_pad_input_w(const float* __restrict__ w, __bf16* __restrict__ out) {
    int r = blockIdx.x;          // 0..127
    int c = threadIdx.x;         // 0..95
    float v = (c < 66) ? w[r * 66 + c] : 0.0f;
    out[r * KIN + c] = (__bf16)v;
}

// ---------------- build xin bf16 [N,96]: embed | time | dwell | zeros ----------------
__global__ void k_build_xin(const int* __restrict__ node_idx,
                            const float* __restrict__ time_feat,
                            const float* __restrict__ dwell_feat,
                            const float* __restrict__ embed,
                            __bf16* __restrict__ xin) {
    int i = blockIdx.x;          // node
    int c = threadIdx.x;         // 0..95
    float v;
    if (c < E_DIM)       v = embed[(size_t)node_idx[i] * E_DIM + c];
    else if (c == 64)    v = time_feat[i];
    else if (c == 65)    v = dwell_feat[i];
    else                 v = 0.0f;
    xin[(size_t)i * KIN + c] = (__bf16)v;
}

// ---------------- WMMA GEMM: Y[N,M] = A[N,K] @ W[M,K]^T (+bias, act) ----------------
// One wave per 16x16 output tile. K multiple of 32, N/M multiples of 16.
// act: 0 = none, 1 = relu.  Cf/Cb optionally null.
__global__ void k_gemm_bf16(const __bf16* __restrict__ A, const __bf16* __restrict__ W,
                            const float* __restrict__ bias,
                            float* __restrict__ Cf, __bf16* __restrict__ Cb,
                            int nrows, int K, int M, int act) {
    int wave = blockIdx.x * (blockDim.x >> 5) + (threadIdx.x >> 5);
    int tilesM = M >> 4;
    int total  = (nrows >> 4) * tilesM;
    if (wave >= total) return;
    int rowT = wave / tilesM;
    int colT = wave % tilesM;
    int lane = threadIdx.x & 31;
    int ln   = lane & 15;
    int h    = lane >> 4;

    const __bf16* arow = A + (size_t)(rowT * 16 + ln) * K;   // A row m = 16-bit A layout lane row
    const __bf16* wrow = W + (size_t)(colT * 16 + ln) * K;   // W row n  (B column n)

    v8f acc = {};
    for (int kb = 0; kb < K; kb += 32) {
        // A 16x32 bf16 layout: lane(m=ln,h) holds K = kb+8h..+7 and kb+16+8h..+7
        v8bf a0 = *(const v8bf*)(arow + kb + h * 8);
        v8bf a1 = *(const v8bf*)(arow + kb + 16 + h * 8);
        // B 32x16 bf16 layout: lane(n=ln,h) holds K = kb+8h..+7 and kb+16+8h..+7
        v8bf b0 = *(const v8bf*)(wrow + kb + h * 8);
        v8bf b1 = *(const v8bf*)(wrow + kb + 16 + h * 8);
        v16bf av, bv;
#pragma unroll
        for (int i = 0; i < 8; ++i) {
            av[i] = a0[i]; av[i + 8] = a1[i];
            bv[i] = b0[i]; bv[i + 8] = b1[i];
        }
        acc = __builtin_amdgcn_wmma_f32_16x16x32_bf16(
            false, av, false, bv, (short)0, acc, false, false);
    }

    int col  = colT * 16 + ln;
    float bb = bias ? bias[col] : 0.0f;
#pragma unroll
    for (int j = 0; j < 8; ++j) {
        int row = rowT * 16 + h * 8 + j;      // C layout: VGPR j -> M = 8h + j, N = ln
        float v = acc[j] + bb;
        if (act == 1) v = v > 0.0f ? v : 0.0f;
        size_t o = (size_t)row * M + col;
        if (Cf) Cf[o] = v;
        if (Cb) Cb[o] = (__bf16)v;
    }
}

// ---------------- edge scatter: m[dst] += t[src]  (f32 atomics, bf16 source) --------
__global__ void k_edge_scatter(const __bf16* __restrict__ t,
                               const int* __restrict__ ei,
                               float* __restrict__ m) {
    long long tid = (long long)blockIdx.x * blockDim.x + threadIdx.x;   // E*32 threads
    int e = (int)(tid >> 5);
    if (e >= N_E) return;
    int c = ((int)tid & 31) * 4;
    int src = ei[e];
    int dst = ei[N_E + e];
    const __bf16* ts = t + (size_t)src * N_H + c;
    float* md = m + (size_t)dst * N_H + c;
#pragma unroll
    for (int k = 0; k < 4; ++k)
        unsafeAtomicAdd(md + k, (float)ts[k]);
}

// ---------------- GRU elementwise update ----------------
__global__ void k_gru(const float* __restrict__ gi, const float* __restrict__ gh,
                      float* __restrict__ x, __bf16* __restrict__ xb) {
    int tid = blockIdx.x * blockDim.x + threadIdx.x;   // N*128 threads
    if (tid >= N_NODES * N_H) return;
    int i = tid >> 7;
    int c = tid & 127;
    size_t g = (size_t)i * (3 * N_H);
    float ir = gi[g + c],          hr = gh[g + c];
    float iz = gi[g + N_H + c],    hz = gh[g + N_H + c];
    float in = gi[g + 2*N_H + c],  hn = gh[g + 2*N_H + c];
    float r = 1.0f / (1.0f + __expf(-(ir + hr)));
    float z = 1.0f / (1.0f + __expf(-(iz + hz)));
    float n = tanhf(in + r * hn);
    float hprev = x[tid];
    float hnew = (1.0f - z) * n + z * hprev;
    x[tid]  = hnew;
    xb[tid] = (__bf16)hnew;
}

// ---------------- attention: only last-position query matters ----------------
// qkv [N, 3H] f32 (bias included). Output ctx bf16 [B, H].
__global__ void k_attn(const float* __restrict__ qkv, __bf16* __restrict__ ctx) {
    int b   = blockIdx.x;
    int tid = threadIdx.x;             // 256 threads
    __shared__ float qs[N_H];
    __shared__ float sc[N_HEADS][N_L];
    __shared__ float inv[N_HEADS];
    const size_t base = (size_t)b * N_L;
    if (tid < N_H) qs[tid] = qkv[(base + N_L - 1) * (3 * N_H) + tid];
    __syncthreads();
    const float scale = 0.17677669529663689f;   // 1/sqrt(32)
    for (int j = tid; j < N_L; j += 256) {
        const float* kr = qkv + (base + j) * (3 * N_H) + N_H;
#pragma unroll
        for (int hd = 0; hd < N_HEADS; ++hd) {
            float s = 0.0f;
            for (int d = 0; d < N_DH; ++d) s += qs[hd * N_DH + d] * kr[hd * N_DH + d];
            sc[hd][j] = s * scale;
        }
    }
    __syncthreads();
    if (tid < N_HEADS) {
        float mx = -1e30f;
        for (int j = 0; j < N_L; ++j) mx = fmaxf(mx, sc[tid][j]);
        float sum = 0.0f;
        for (int j = 0; j < N_L; ++j) { float e = __expf(sc[tid][j] - mx); sc[tid][j] = e; sum += e; }
        inv[tid] = 1.0f / sum;
    }
    __syncthreads();
    if (tid < N_H) {
        int hd = tid >> 5;
        float acc = 0.0f;
        for (int j = 0; j < N_L; ++j)
            acc += sc[hd][j] * qkv[(base + j) * (3 * N_H) + 2 * N_H + hd * N_DH + (tid & 31)];
        ctx[(size_t)b * N_H + tid] = (__bf16)(acc * inv[hd]);
    }
}

// ---------------- host-side orchestration ----------------
static inline int gemm_blocks(int nrows, int M) {
    int waves = (nrows >> 4) * (M >> 4);
    return (waves + 7) / 8;        // 8 waves per 256-thread block
}

extern "C" void kernel_launch(void* const* d_in, const int* in_sizes, int n_in,
                              void* d_out, int out_size, void* d_ws, size_t ws_size,
                              hipStream_t stream) {
    const int*   node_idx   = (const int*)  d_in[0];
    const float* time_feat  = (const float*)d_in[1];
    const float* dwell_feat = (const float*)d_in[2];
    const int*   edge_index = (const int*)  d_in[3];
    // d_in[4] = batch (sorted repeat(arange(B), L)) -> structure known statically
    const float* embed      = (const float*)d_in[5];
    const float* input_w    = (const float*)d_in[6];
    const float* input_b    = (const float*)d_in[7];
    const float* ggnn_w     = (const float*)d_in[8];
    const float* gru_wih    = (const float*)d_in[9];
    const float* gru_whh    = (const float*)d_in[10];
    const float* gru_bih    = (const float*)d_in[11];
    const float* gru_bhh    = (const float*)d_in[12];
    const float* attn_in_w  = (const float*)d_in[13];
    const float* attn_in_b  = (const float*)d_in[14];
    const float* attn_out_w = (const float*)d_in[15];
    const float* attn_out_b = (const float*)d_in[16];
    const float* lin_w      = (const float*)d_in[17];
    const float* lin_b      = (const float*)d_in[18];
    float* out = (float*)d_out;

    // ---- carve workspace ----
    char* ws = (char*)d_ws;
    size_t off = 0;
    auto alloc = [&](size_t bytes) -> char* {
        char* p = ws + off;
        off += (bytes + 255) & ~(size_t)255;
        return p;
    };
    __bf16* xin_b   = (__bf16*)alloc((size_t)N_NODES * KIN * 2);
    float*  x_f     = (float*) alloc((size_t)N_NODES * N_H * 4);
    __bf16* x_b     = (__bf16*)alloc((size_t)N_NODES * N_H * 2);
    __bf16* t_b     = (__bf16*)alloc((size_t)N_NODES * N_H * 2);
    float*  m_f     = (float*) alloc((size_t)N_NODES * N_H * 4);
    __bf16* m_b     = (__bf16*)alloc((size_t)N_NODES * N_H * 2);
    float*  gi_f    = (float*) alloc((size_t)N_NODES * 3 * N_H * 4);  // also reused for qkv
    float*  gh_f    = (float*) alloc((size_t)N_NODES * 3 * N_H * 4);
    __bf16* ctx_b   = (__bf16*)alloc((size_t)N_B * N_H * 2);
    __bf16* fin_b   = (__bf16*)alloc((size_t)N_B * N_H * 2);
    // bf16 weights
    __bf16* w_in_b  = (__bf16*)alloc((size_t)N_H * KIN * 2);
    __bf16* w_gg_b  = (__bf16*)alloc((size_t)2 * N_H * N_H * 2);
    __bf16* w_ih_b  = (__bf16*)alloc((size_t)3 * N_H * N_H * 2);
    __bf16* w_hh_b  = (__bf16*)alloc((size_t)3 * N_H * N_H * 2);
    __bf16* w_ai_b  = (__bf16*)alloc((size_t)3 * N_H * N_H * 2);
    __bf16* w_ao_b  = (__bf16*)alloc((size_t)N_H * N_H * 2);
    __bf16* w_lin_b = (__bf16*)alloc((size_t)N_CLASSES * N_H * 2);

    auto cvt = [&](const float* src, __bf16* dst, int n) {
        k_cvt_bf16<<<(n + 255) / 256, 256, 0, stream>>>(src, dst, n);
    };

    // ---- stage weights to bf16 ----
    k_pad_input_w<<<N_H, KIN, 0, stream>>>(input_w, w_in_b);
    cvt(ggnn_w,     w_gg_b,  2 * N_H * N_H);
    cvt(gru_wih,    w_ih_b,  3 * N_H * N_H);
    cvt(gru_whh,    w_hh_b,  3 * N_H * N_H);
    cvt(attn_in_w,  w_ai_b,  3 * N_H * N_H);
    cvt(attn_out_w, w_ao_b,  N_H * N_H);
    cvt(lin_w,      w_lin_b, N_CLASSES * N_H);

    // ---- input features + input linear (relu) ----
    k_build_xin<<<N_NODES, KIN, 0, stream>>>(node_idx, time_feat, dwell_feat, embed, xin_b);
    k_gemm_bf16<<<gemm_blocks(N_NODES, N_H), 256, 0, stream>>>(
        xin_b, w_in_b, input_b, x_f, x_b, N_NODES, KIN, N_H, 1);

    // ---- GGNN layers ----
    for (int l = 0; l < 2; ++l) {
        // t = x @ ggnn_w[l]^T  (bf16 out only)
        k_gemm_bf16<<<gemm_blocks(N_NODES, N_H), 256, 0, stream>>>(
            x_b, w_gg_b + (size_t)l * N_H * N_H, nullptr, nullptr, t_b,
            N_NODES, N_H, N_H, 0);
        // m = segment_sum(t[src] -> dst)
        hipMemsetAsync(m_f, 0, (size_t)N_NODES * N_H * 4, stream);
        {
            long long thr = (long long)N_E * 32;
            k_edge_scatter<<<(int)((thr + 255) / 256), 256, 0, stream>>>(t_b, edge_index, m_f);
        }
        cvt(m_f, m_b, N_NODES * N_H);
        // gi = m @ wih^T + bih ; gh = x @ whh^T + bhh
        k_gemm_bf16<<<gemm_blocks(N_NODES, 3 * N_H), 256, 0, stream>>>(
            m_b, w_ih_b, gru_bih, gi_f, nullptr, N_NODES, N_H, 3 * N_H, 0);
        k_gemm_bf16<<<gemm_blocks(N_NODES, 3 * N_H), 256, 0, stream>>>(
            x_b, w_hh_b, gru_bhh, gh_f, nullptr, N_NODES, N_H, 3 * N_H, 0);
        // GRU update (in-place x)
        k_gru<<<(N_NODES * N_H + 255) / 256, 256, 0, stream>>>(gi_f, gh_f, x_f, x_b);
    }

    // ---- attention: qkv for all nodes, query only at last position per session ----
    k_gemm_bf16<<<gemm_blocks(N_NODES, 3 * N_H), 256, 0, stream>>>(
        x_b, w_ai_b, attn_in_b, gi_f /* qkv */, nullptr, N_NODES, N_H, 3 * N_H, 0);
    k_attn<<<N_B, 256, 0, stream>>>(gi_f, ctx_b);
    // out projection on the B last-position rows
    k_gemm_bf16<<<gemm_blocks(N_B, N_H), 256, 0, stream>>>(
        ctx_b, w_ao_b, attn_out_b, nullptr, fin_b, N_B, N_H, N_H, 0);

    // ---- final classifier: [B, 10000] ----
    k_gemm_bf16<<<gemm_blocks(N_B, N_CLASSES), 256, 0, stream>>>(
        fin_b, w_lin_b, lin_b, out, nullptr, N_B, N_H, N_CLASSES, 0);
}
